// SnSymmetryBasedModel_11922829214358
// MI455X (gfx1250) — matrix operand order repaired and checked by reading, not compiled, not run on hardware
//
#include <hip/hip_runtime.h>

#define N_NODES 1048576
#define D0      64
#define H       128
#define NGRAPH  4096
#define BN_EPS  1e-5f

typedef __attribute__((ext_vector_type(16))) __bf16 v16bf;
typedef __attribute__((ext_vector_type(8)))  float  v8f;

__device__ __forceinline__ unsigned short f2bf(float f) {
  unsigned u = __float_as_uint(f);
  u = (u + 0x7fffu + ((u >> 16) & 1u)) >> 16;   // round-to-nearest-even
  return (unsigned short)u;
}
__device__ __forceinline__ float bf2f(unsigned short h) {
  return __uint_as_float(((unsigned)h) << 16);
}

union FragBF { uint4 q[2]; v16bf v; };

// ---------------------------------------------------------------------------
// Fused  Out[M,128] = (relu(A[M,K] @ W1 + b1)) @ W2 + b2  (+ addR[bidx[row]])
// A is f32 row-major; W1t/W2t are bf16 stored TRANSPOSED: Wt[n][k].
// Block = 256 threads (8 waves), 128 rows per block, bf16 WMMA, f32 accum.
// All 8 B-fragments of a k-step are preloaded into registers so the 8 WMMAs
// drain the DS counter progressively instead of waiting per-instruction.
// ---------------------------------------------------------------------------
template <int K>
__global__ __launch_bounds__(256) void fused_mlp_kernel(
    const float* __restrict__ A, const unsigned short* __restrict__ W1t,
    const float* __restrict__ b1, const unsigned short* __restrict__ W2t,
    const float* __restrict__ b2, float* __restrict__ Out,
    const float* __restrict__ addR, const int* __restrict__ bidx) {
  constexpr int SA = K + 8;   // padded LDS row stride (bf16 elems)
  constexpr int ST = H + 8;
  extern __shared__ unsigned short smem[];
  unsigned short* sA  = smem;              // 128 x SA   (A panel, bf16)
  unsigned short* sT  = sA + 128 * SA;     // 128 x ST   (mid activation)
  unsigned short* sW1 = sT + 128 * ST;     // 128 x SA   (W1t)
  unsigned short* sW2 = sW1 + 128 * SA;    // 128 x ST   (W2t)

  const int tid = threadIdx.x;
  const int lane = tid & 31;
  const int wave = tid >> 5;
  const int rowBase = blockIdx.x * 128;
  const int waveRow = wave * 16;

  // ---- stage A panel: f32 -> bf16 into LDS (2 threads per row) ----
  {
    int r = tid >> 1;
    int kOff = (tid & 1) * (K / 2);
    const float* src = A + (size_t)(rowBase + r) * K + kOff;
    unsigned short* dst = sA + r * SA + kOff;
    __builtin_prefetch(A + (size_t)(rowBase + r + 128) * K + kOff, 0, 1);
#pragma unroll
    for (int k = 0; k < K / 2; k += 4) {
      float4 v = *(const float4*)(src + k);
      uint2 p;
      p.x = (unsigned)f2bf(v.x) | ((unsigned)f2bf(v.y) << 16);
      p.y = (unsigned)f2bf(v.z) | ((unsigned)f2bf(v.w) << 16);
      *(uint2*)(dst + k) = p;
    }
  }
  // ---- stage W1t ----
  {
    int n = tid >> 1;
    int kOff = (tid & 1) * (K / 2);
    const unsigned short* src = W1t + (size_t)n * K + kOff;
    unsigned short* dst = sW1 + n * SA + kOff;
#pragma unroll
    for (int k = 0; k < K / 2; k += 8) *(uint4*)(dst + k) = *(const uint4*)(src + k);
  }
  // ---- stage W2t ----
  {
    int n = tid >> 1;
    int kOff = (tid & 1) * 64;
    const unsigned short* src = W2t + (size_t)n * H + kOff;
    unsigned short* dst = sW2 + n * ST + kOff;
#pragma unroll
    for (int k = 0; k < 64; k += 8) *(uint4*)(dst + k) = *(const uint4*)(src + k);
  }
  __syncthreads();

  const v8f z8 = {0.f, 0.f, 0.f, 0.f, 0.f, 0.f, 0.f, 0.f};
  v8f acc[8];
#pragma unroll
  for (int nt = 0; nt < 8; ++nt) acc[nt] = z8;

  // ================= GEMM1: t = A @ W1 =================
#pragma unroll
  for (int kt = 0; kt < K / 32; ++kt) {
    FragBF fa;
    const int arow = waveRow + (lane & 15);
    const int kbA = kt * 32 + ((lane >> 4) << 3);   // lanes<16: K 0..7/16..23
    fa.q[0] = *(const uint4*)(sA + arow * SA + kbA);
    fa.q[1] = *(const uint4*)(sA + arow * SA + kbA + 16);
    const int kbB = kt * 32 + ((lane >> 4) << 3);   // lanes<16: K 0..7/16..23
    FragBF fb[8];
#pragma unroll
    for (int nt = 0; nt < 8; ++nt) {
      const int bn = nt * 16 + (lane & 15);
      fb[nt].q[0] = *(const uint4*)(sW1 + bn * SA + kbB);
      fb[nt].q[1] = *(const uint4*)(sW1 + bn * SA + kbB + 16);
    }
#pragma unroll
    for (int nt = 0; nt < 8; ++nt)
      acc[nt] = __builtin_amdgcn_wmma_f32_16x16x32_bf16(
          false, fa.v, false, fb[nt].v, (short)0, acc[nt], false, false);
  }

  // ---- epilogue1: +b1, ReLU, bf16 -> sT (each wave writes only its rows) ---
  {
    const int rB = waveRow + ((lane >> 4) << 3);
#pragma unroll
    for (int nt = 0; nt < 8; ++nt) {
      const int col = nt * 16 + (lane & 15);
      const float bb = b1[col];
#pragma unroll
      for (int j = 0; j < 8; ++j) {
        float v = acc[nt][j] + bb;
        v = v > 0.f ? v : 0.f;
        sT[(rB + j) * ST + col] = f2bf(v);
      }
    }
  }
  __syncthreads();

  // ================= GEMM2: h = t @ W2 =================
#pragma unroll
  for (int nt = 0; nt < 8; ++nt) acc[nt] = z8;
#pragma unroll
  for (int kt = 0; kt < H / 32; ++kt) {
    FragBF fa;
    const int arow = waveRow + (lane & 15);
    const int kbA = kt * 32 + ((lane >> 4) << 3);
    fa.q[0] = *(const uint4*)(sT + arow * ST + kbA);
    fa.q[1] = *(const uint4*)(sT + arow * ST + kbA + 16);
    const int kbB = kt * 32 + ((lane >> 4) << 3);
    FragBF fb[8];
#pragma unroll
    for (int nt = 0; nt < 8; ++nt) {
      const int bn = nt * 16 + (lane & 15);
      fb[nt].q[0] = *(const uint4*)(sW2 + bn * ST + kbB);
      fb[nt].q[1] = *(const uint4*)(sW2 + bn * ST + kbB + 16);
    }
#pragma unroll
    for (int nt = 0; nt < 8; ++nt)
      acc[nt] = __builtin_amdgcn_wmma_f32_16x16x32_bf16(
          false, fa.v, false, fb[nt].v, (short)0, acc[nt], false, false);
  }

  // ---- epilogue2: +b2 (+ rho broadcast), store f32 ----
  {
    const int rB = waveRow + ((lane >> 4) << 3);
    int gidx[8];
    if (addR) {
#pragma unroll
      for (int j = 0; j < 8; ++j) gidx[j] = bidx[rowBase + rB + j];
    }
#pragma unroll
    for (int nt = 0; nt < 8; ++nt) {
      const int col = nt * 16 + (lane & 15);
      const float bb = b2[col];
#pragma unroll
      for (int j = 0; j < 8; ++j) {
        float v = acc[nt][j] + bb;
        if (addR) v += addR[(size_t)gidx[j] * H + col];
        Out[(size_t)(rowBase + rB + j) * H + col] = v;
      }
    }
  }
}

// ---------------------------------------------------------------------------
__global__ void zero_kernel(float* __restrict__ p, int n) {
  int t = blockIdx.x * 256 + threadIdx.x;
  if (t < n) p[t] = 0.f;
}

// transpose + f32->bf16:  Wt[n*K + k] = bf16(W[k*Nc + n]), Nc = H
__global__ void wtrans_kernel(const float* __restrict__ W,
                              unsigned short* __restrict__ Wt, int K, int Nc) {
  int t = blockIdx.x * 256 + threadIdx.x;
  if (t >= K * Nc) return;
  int n = t / K, k = t % K;
  Wt[t] = f2bf(W[(size_t)k * Nc + n]);
}

// segment sum: s[g, :] += x[n, :] ; sorted bidx, f32 atomics
__global__ void segsum_kernel(const float* __restrict__ x,
                              const int* __restrict__ bidx,
                              float* __restrict__ s, int n_rows, int d) {
  long long t = (long long)blockIdx.x * 256 + threadIdx.x;
  const int per = d >> 2;
  if (t >= (long long)n_rows * per) return;
  int n = (int)(t / per);
  int f = (int)(t % per) * 4;
  int g = bidx[n];
  float4 v = *(const float4*)(x + (size_t)n * d + f);
  float* dst = s + (size_t)g * d + f;
  atomicAdd(dst + 0, v.x);
  atomicAdd(dst + 1, v.y);
  atomicAdd(dst + 2, v.z);
  atomicAdd(dst + 3, v.w);
}

// per-channel sum / sumsq over N rows
__global__ void bn_stats_kernel(const float* __restrict__ h, float* __restrict__ sums,
                                float* __restrict__ sumq, int rowsPerBlock) {
  __shared__ float ls[256], lq[256];
  const int tid = threadIdx.x;
  const int col = tid & 127, half = tid >> 7;
  const size_t base = (size_t)blockIdx.x * rowsPerBlock;
  float s = 0.f, q = 0.f;
  for (int r = half; r < rowsPerBlock; r += 2) {
    float v = h[(base + r) * H + col];
    s += v; q += v * v;
  }
  ls[tid] = s; lq[tid] = q;
  __syncthreads();
  if (half == 0) {
    atomicAdd(&sums[col], ls[tid] + ls[tid + 128]);
    atomicAdd(&sumq[col], lq[tid] + lq[tid + 128]);
  }
}

// xout = (h - mean)*rsqrt(var+eps)*gamma + beta (+ residual)
__global__ void bn_apply_kernel(const float* __restrict__ h, const float* __restrict__ xres,
                                const float* __restrict__ sums, const float* __restrict__ sumq,
                                const float* __restrict__ gamma, const float* __restrict__ beta,
                                float* __restrict__ xout, float invN) {
  size_t i = ((size_t)blockIdx.x * 256 + threadIdx.x) * 4;
  if (i >= (size_t)N_NODES * H) return;
  int c = (int)(i & (H - 1));
  float4 hv = *(const float4*)(h + i);
  float hvv[4] = {hv.x, hv.y, hv.z, hv.w}, o[4];
#pragma unroll
  for (int j = 0; j < 4; ++j) {
    float m = sums[c + j] * invN;
    float var = sumq[c + j] * invN - m * m;
    o[j] = (hvv[j] - m) * rsqrtf(var + BN_EPS) * gamma[c + j] + beta[c + j];
  }
  float4 ov = {o[0], o[1], o[2], o[3]};
  if (xres) {
    float4 xv = *(const float4*)(xres + i);
    ov.x += xv.x; ov.y += xv.y; ov.z += xv.z; ov.w += xv.w;
  }
  *(float4*)(xout + i) = ov;
}

// final pooling head: out[g] = b2 + sum_c relu(b1[c] + s2[g,:]·W1[:,c]) * W2[c]
__global__ void pool_out_kernel(const float* __restrict__ s2,
                                const unsigned short* __restrict__ W1t,
                                const float* __restrict__ b1, const float* __restrict__ W2,
                                const float* __restrict__ b2, float* __restrict__ out) {
  __shared__ float sv[H];
  __shared__ float red[H];
  const int g = blockIdx.x, c = threadIdx.x;
  sv[c] = s2[(size_t)g * H + c];
  __syncthreads();
  float a = b1[c];
  for (int k = 0; k < H; ++k) a += sv[k] * bf2f(W1t[c * H + k]);
  a = a > 0.f ? a : 0.f;
  red[c] = a * W2[c];
  __syncthreads();
  for (int off = 64; off > 0; off >>= 1) {
    if (c < off) red[c] += red[c + off];
    __syncthreads();
  }
  if (c == 0) out[g] = red[0] + b2[0];
}

// ---------------------------------------------------------------------------
extern "C" void kernel_launch(void* const* d_in, const int* in_sizes, int n_in,
                              void* d_out, int out_size, void* d_ws, size_t ws_size,
                              hipStream_t stream) {
  (void)in_sizes; (void)n_in; (void)out_size; (void)ws_size;
  const float* x0 = (const float*)d_in[0];
  const int* bidx = (const int*)d_in[1];

  // jax pytree flatten order (dict keys sorted; 'W' < 'b'):
  // per layer: bn_beta, bn_gamma, phi0.W, phi0.b, phi1.W, phi1.b, rho0.W, rho0.b, rho1.W, rho1.b
  struct LP { const float *beta,*gamma,*pW1,*pB1,*pW2,*pB2,*rW1,*rB1,*rW2,*rB2; };
  LP L[3];
  const int Ks[3] = {D0, H, H};
  int base = 2;
  for (int i = 0; i < 3; ++i) {
    L[i].beta  = (const float*)d_in[base + 0];
    L[i].gamma = (const float*)d_in[base + 1];
    L[i].pW1 = (const float*)d_in[base + 2]; L[i].pB1 = (const float*)d_in[base + 3];
    L[i].pW2 = (const float*)d_in[base + 4]; L[i].pB2 = (const float*)d_in[base + 5];
    L[i].rW1 = (const float*)d_in[base + 6]; L[i].rB1 = (const float*)d_in[base + 7];
    L[i].rW2 = (const float*)d_in[base + 8]; L[i].rB2 = (const float*)d_in[base + 9];
    base += 10;
  }
  const float* ppW1 = (const float*)d_in[base + 0]; const float* ppB1 = (const float*)d_in[base + 1];
  const float* ppW2 = (const float*)d_in[base + 2]; const float* ppB2 = (const float*)d_in[base + 3];
  const float* prW1 = (const float*)d_in[base + 4]; const float* prB1 = (const float*)d_in[base + 5];
  const float* prW2 = (const float*)d_in[base + 6]; const float* prB2 = (const float*)d_in[base + 7];

  const size_t NH = (size_t)N_NODES * H;
  char* ws = (char*)d_ws;
  float* bufA = (float*)ws;                          // [N,128] activations
  float* bufB = (float*)(ws + NH * 4);               // [N,128] MLP output
  float* sS   = (float*)(ws + 2 * NH * 4);           // [G,128] segment sums
  float* sR   = sS + (size_t)NGRAPH * H;             // [G,128] rho output
  float* sums = sR + (size_t)NGRAPH * H;             // [128]
  float* sumq = sums + H;                            // [128]
  unsigned short* wb = (unsigned short*)(sumq + H);  // bf16 transposed weights

  unsigned short *pW1t[3], *pW2t[3], *rW1t[3], *rW2t[3];
  for (int i = 0; i < 3; ++i) {
    pW1t[i] = wb; wb += (size_t)Ks[i] * H;
    pW2t[i] = wb; wb += (size_t)H * H;
    rW1t[i] = wb; wb += (size_t)Ks[i] * H;
    rW2t[i] = wb; wb += (size_t)H * H;
  }
  unsigned short* ppW1t = wb; wb += (size_t)H * H;
  unsigned short* ppW2t = wb; wb += (size_t)H * H;
  unsigned short* prW1t = wb; wb += (size_t)H * H;

  auto wt = [&](const float* W, unsigned short* Wt, int K) {
    int n = K * H;
    wtrans_kernel<<<(n + 255) / 256, 256, 0, stream>>>(W, Wt, K, H);
  };
  for (int i = 0; i < 3; ++i) {
    wt(L[i].pW1, pW1t[i], Ks[i]); wt(L[i].pW2, pW2t[i], H);
    wt(L[i].rW1, rW1t[i], Ks[i]); wt(L[i].rW2, rW2t[i], H);
  }
  wt(ppW1, ppW1t, H); wt(ppW2, ppW2t, H); wt(prW1, prW1t, H);

  const float* xcur = x0;
  for (int i = 0; i < 3; ++i) {
    const int K = Ks[i];
    const size_t shb = (size_t)128 * (2 * (K + 8) + 2 * (H + 8)) * 2;

    // segment sum of current activations
    zero_kernel<<<(NGRAPH * K + 255) / 256, 256, 0, stream>>>(sS, NGRAPH * K);
    {
      long long tot = (long long)N_NODES * (K / 4);
      segsum_kernel<<<(unsigned)((tot + 255) / 256), 256, 0, stream>>>(xcur, bidx, sS, N_NODES, K);
    }
    // rho MLP on G rows only (256x less work than broadcasting first)
    // phi MLP on N rows, fused add of rho[batch_idx]
    if (K == 64) {
      fused_mlp_kernel<64><<<NGRAPH / 128, 256, shb, stream>>>(
          sS, rW1t[i], L[i].rB1, rW2t[i], L[i].rB2, sR, nullptr, nullptr);
      fused_mlp_kernel<64><<<N_NODES / 128, 256, shb, stream>>>(
          xcur, pW1t[i], L[i].pB1, pW2t[i], L[i].pB2, bufB, sR, bidx);
    } else {
      fused_mlp_kernel<128><<<NGRAPH / 128, 256, shb, stream>>>(
          sS, rW1t[i], L[i].rB1, rW2t[i], L[i].rB2, sR, nullptr, nullptr);
      fused_mlp_kernel<128><<<N_NODES / 128, 256, shb, stream>>>(
          xcur, pW1t[i], L[i].pB1, pW2t[i], L[i].pB2, bufB, sR, bidx);
    }
    // batch norm + residual (in-place into bufA)
    zero_kernel<<<1, 256, 0, stream>>>(sums, 2 * H);
    bn_stats_kernel<<<2048, 256, 0, stream>>>(bufB, sums, sumq, N_NODES / 2048);
    bn_apply_kernel<<<(unsigned)(NH / 4 / 256), 256, 0, stream>>>(
        bufB, (i == 0 ? nullptr : bufA), sums, sumq, L[i].gamma, L[i].beta, bufA,
        1.0f / (float)N_NODES);
    xcur = bufA;
  }

  // pooling: phi over N, segment-sum, tiny rho head -> [G]
  {
    const size_t shb = (size_t)128 * (2 * (H + 8) + 2 * (H + 8)) * 2;
    fused_mlp_kernel<128><<<N_NODES / 128, 256, shb, stream>>>(
        bufA, ppW1t, ppB1, ppW2t, ppB2, bufB, nullptr, nullptr);
    zero_kernel<<<(NGRAPH * H + 255) / 256, 256, 0, stream>>>(sS, NGRAPH * H);
    long long tot = (long long)N_NODES * (H / 4);
    segsum_kernel<<<(unsigned)((tot + 255) / 256), 256, 0, stream>>>(bufB, bidx, sS, N_NODES, H);
    pool_out_kernel<<<NGRAPH, H, 0, stream>>>(sS, prW1t, prB1, prW2, prB2, (float*)d_out);
  }
}
